// Decoder_14431090114949
// MI455X (gfx1250) — matrix operand order repaired and checked by reading, not compiled, not run on hardware
//
#include <hip/hip_runtime.h>
#include <hip/hip_bf16.h>
#include <math.h>

// ---------------------------------------------------------------------------
// Video Pixel Network decoder stack (4 RMBs) on gfx1250 via bf16 WMMA.
//   activations: bf16, layout [pixel][channel] (channel-fast)
//   weights:     device-packed bf16 B-fragments (one 32B load per lane,
//                constant stride walk -> immediate-offset loads)
//   accumulate:  f32 via v_wmma_f32_16x16x32_bf16
//   MU convs:    LDS-staged halo tiles (ds_load_b128 A-fragments, zeroed halo)
// ---------------------------------------------------------------------------

typedef __bf16 bf16_t;
typedef __attribute__((ext_vector_type(16))) __bf16 v16bf;
typedef __attribute__((ext_vector_type(8)))  float  v8f;
typedef __attribute__((ext_vector_type(4)))  float  vf4;

#define NPIX   131072   // 8 images * 128*128
#define IMG_HW 16384
#define IMG_W  128
#define IMG_H  128

__device__ __forceinline__ v8f wmma_bf16(v16bf a, v16bf b, v8f c) {
  return __builtin_amdgcn_wmma_f32_16x16x32_bf16(false, a, false, b, (short)0, c,
                                                 false, false);
}

// A fragment (16 px x 32 ch bf16). Per ISA layout: lanes 0-15 row M=lane hold
// K{0..7,16..23}; lanes 16-31 row M=lane-16 hold K{8..15,24..31}. Caller passes
// p already offset by the lane's K-segment base (0 or 8 halfs); we load 8 halfs
// at p and 8 halfs at p+16. Works for both global and LDS pointers.
__device__ __forceinline__ v16bf load_a_frag(const bf16_t* p) {
  union { v16bf v; vf4 f[2]; } u;
  u.f[0] = *reinterpret_cast<const vf4*>(p);
  u.f[1] = *reinterpret_cast<const vf4*>(p + 16);
  return u.v;
}

// B fragment: lane's 16 halfs stored contiguously by the pack kernel.
__device__ __forceinline__ v16bf load_b_frag(const bf16_t* __restrict__ p) {
  union { v16bf v; vf4 f[2]; } u;
  u.f[0] = *reinterpret_cast<const vf4*>(p);
  u.f[1] = *reinterpret_cast<const vf4*>(p + 8);
  return u.v;
}

__device__ __forceinline__ float sigmoidf_(float x) { return 1.f / (1.f + __expf(-x)); }

// ---------------------------------------------------------------------------
// Weight packing: OIHW f32 -> bf16 fragments.
// dst[((tap*KC + kc)*NT + ntile)*512 + lane*16 + j] =
//     W[o = ntile*16 + (lane&15)][i = kc*32 + ((j&8)?16:0) + ((lane&16)?8:0) + (j&7)][tap]
// Taps (kh,kw): 0:(0,0) 1:(0,1) 2:(0,2) 3:(1,0) 4:(1,1)  (PixelCNN-causal set)
// ---------------------------------------------------------------------------
__global__ void pack_w_kernel(const float* __restrict__ w, bf16_t* __restrict__ dst,
                              int O, int Isrc, int Kpad, int is3x3, int zero_center,
                              int total) {
  int tid = blockIdx.x * blockDim.x + threadIdx.x;
  if (tid >= total) return;
  int j    = tid & 15;
  int lane = (tid >> 4) & 31;
  int rest = tid >> 9;
  int NT = O >> 4, KC = Kpad >> 5;
  int ntile = rest % NT; rest /= NT;
  int kc  = rest % KC;
  int tap = rest / KC;
  int o  = ntile * 16 + (lane & 15);
  int kl = ((j & 8) ? 16 : 0) + ((lane & 16) ? 8 : 0) + (j & 7);
  int i  = kc * 32 + kl;
  float v = 0.f;
  if (i < Isrc) {
    if (is3x3) {
      int kh = (tap < 3) ? 0 : 1;
      int kw = (tap < 3) ? tap : (tap - 3);
      v = w[((o * Isrc + i) * 3 + kh) * 3 + kw];
      if (zero_center && tap == 4) v = 0.f;   // mask 'A' excludes center pixel
    } else {
      v = w[o * Isrc + i];
    }
  }
  dst[tid] = (bf16_t)v;
}

// inputs (8,32,128,128) f32 NCHW + targets (8,1,128,128) ->
//   xin64[pix][64] bf16 (ch 0-31 = x, 32 = frame, 33-63 = 0)  and
//   x32  [pix][32] bf16 (residual carrier)
__global__ void pack_input_kernel(const float* __restrict__ xin,
                                  const float* __restrict__ tgt,
                                  bf16_t* __restrict__ xin64,
                                  bf16_t* __restrict__ x32) {
  int tid = blockIdx.x * blockDim.x + threadIdx.x;
  if (tid >= NPIX * 64) return;
  int pix = tid >> 6, c = tid & 63;
  int img = pix >> 14, yx = pix & (IMG_HW - 1);
  float v = 0.f;
  if (c < 32)       v = xin[((size_t)img * 32 + c) * IMG_HW + yx];
  else if (c == 32) v = tgt[(size_t)img * IMG_HW + yx];
  xin64[tid] = (bf16_t)v;
  if (c < 32) x32[(size_t)pix * 32 + c] = (bf16_t)v;
}

// ---------------------------------------------------------------------------
// 1x1 conv as GEMM. One wave = (16-pixel tile, NG consecutive 16-out-ch tiles):
// the A fragment is loaded once and reused for NG WMMAs. NNT = total 16-ch
// tiles, NK = K/32; all loops unroll, fragment offsets become immediates.
// MODE (compile-time, branch-free epilogues):
//   0: store bf16 [pix][NNT*16]
//   1: store bf16 [pix][NNT*16] + residual add
//   2: store f32 NCHW (final 256-ch logits)
// ---------------------------------------------------------------------------
template <int NG, int NNT, int NK, int MODE>
__global__ void __launch_bounds__(256)
gemm1x1_kernel(const bf16_t* __restrict__ A, int Cin,
               const bf16_t* __restrict__ Wp, const float* __restrict__ bias,
               const bf16_t* __restrict__ resid,
               bf16_t* __restrict__ outb, float* __restrict__ outf,
               int ntasks) {
  int wave = blockIdx.x * (blockDim.x >> 5) + (threadIdx.x >> 5);
  if (wave >= ntasks) return;
  const int ngrp = NNT / NG;
  int tile = wave / ngrp;
  int g0   = (wave - tile * ngrp) * NG;   // first 16-ch tile handled
  int lane = threadIdx.x & 31;
  size_t pix0 = (size_t)tile * 16;
  v8f acc[NG];
  #pragma unroll
  for (int g = 0; g < NG; ++g) acc[g] = (v8f){};
  const bf16_t* ap = A + (pix0 + (lane & 15)) * Cin + ((lane & 16) ? 8 : 0);
  const bf16_t* bp = Wp + (size_t)g0 * 512 + lane * 16;
  #pragma unroll
  for (int kc = 0; kc < NK; ++kc) {
    v16bf a = load_a_frag(ap + kc * 32);
    v16bf bw[NG];
    #pragma unroll
    for (int g = 0; g < NG; ++g) bw[g] = load_b_frag(bp + kc * (NNT * 512) + g * 512);
    #pragma unroll
    for (int g = 0; g < NG; ++g) acc[g] = wmma_bf16(a, bw[g], acc[g]);
  }
  int n = lane & 15, mb = (lane >> 4) * 8;
  const int Nch = NNT * 16;
  #pragma unroll
  for (int g = 0; g < NG; ++g) {
    int ch = (g0 + g) * 16 + n;
    float bb = bias[ch];
    #pragma unroll
    for (int r = 0; r < 8; ++r) {
      size_t p = pix0 + mb + r;
      float v = acc[g][r] + bb;
      if (MODE == 1) v += (float)resid[p * Nch + ch];
      if (MODE == 2) {
        size_t img = p >> 14, yx = p & (IMG_HW - 1);
        outf[(img * 256 + ch) * IMG_HW + yx] = v;
      } else {
        outb[p * Nch + ch] = (bf16_t)v;
      }
    }
  }
}

// ---------------------------------------------------------------------------
// Fused Multiplicative Unit: 4 masked 3x3 convs (shared A fragment) + gating,
// LDS-staged. One block = 32 pixels of one image row (2 tiles x 4 ch-tiles =
// 8 waves). Stage rows {y-1, y}, x in [x0-1, x0+32] (34 slots) with halo
// zero-filled -> hot loop has no boundary predication; A-fragments come from
// ds_load_b128 and gating h reads come from LDS. Weight fragments are walked
// with a constant 2048-half stride (immediate-offset global loads), and the
// four B fragments live in distinct registers so their loads overlap.
//   hout = sigmoid(c1) * tanh(sigmoid(c2)*h + sigmoid(c3)*tanh(c4))
// ---------------------------------------------------------------------------
#define LDS_PITCH 72            // halfs per pixel slot (64 ch + 8 pad)
#define LDS_SLOTS (2 * 34)

template <int NTAPS>
__global__ void __launch_bounds__(256)
mu_kernel(const bf16_t* __restrict__ hin, bf16_t* __restrict__ hout,
          const bf16_t* __restrict__ w1p, const bf16_t* __restrict__ w2p,
          const bf16_t* __restrict__ w3p, const bf16_t* __restrict__ w4p,
          const float* __restrict__ b1, const float* __restrict__ b2,
          const float* __restrict__ b3, const float* __restrict__ b4) {
  __shared__ bf16_t lds[LDS_SLOTS * LDS_PITCH];

  int bid  = blockIdx.x;
  int row  = bid >> 2;             // global row: img*128 + y, in [0, 1024)
  int xs4  = bid & 3;
  int y    = row & (IMG_H - 1);
  int x0   = xs4 * 32;
  size_t rowpix = (size_t)row * IMG_W;
  int tid = threadIdx.x;

  // ---- cooperative stage: 68 slots x 4 segments of 16 halfs (32B) ----
  for (int it = tid; it < LDS_SLOTS * 4; it += 256) {
    int slot = it >> 2, seg = it & 3;
    int r  = slot >= 34 ? 1 : 0;     // 0: row y-1, 1: row y
    int xi = slot - r * 34;          // 0..33 -> x = x0-1+xi
    int xx = x0 - 1 + xi;
    int yy = y + r - 1;
    bf16_t* dst = lds + (size_t)slot * LDS_PITCH + seg * 16;
    if (((unsigned)xx < (unsigned)IMG_W) && ((unsigned)yy < (unsigned)IMG_H)) {
      const bf16_t* src = hin + (rowpix + (size_t)(r - 1) * IMG_W + xx) * 64 + seg * 16;
      *reinterpret_cast<vf4*>(dst)     = *reinterpret_cast<const vf4*>(src);
      *reinterpret_cast<vf4*>(dst + 8) = *reinterpret_cast<const vf4*>(src + 8);
    } else {
      vf4 z = {0.f, 0.f, 0.f, 0.f};
      *reinterpret_cast<vf4*>(dst)     = z;
      *reinterpret_cast<vf4*>(dst + 8) = z;
    }
  }
  __syncthreads();

  // ---- per-wave WMMA: wave = (sub-tile of 16 px, 16-out-ch tile) ----
  int wave = tid >> 5, lane = tid & 31;
  int sub = wave >> 2, ntile = wave & 3;
  int m = lane & 15;
  int kseg = (lane & 16) ? 8 : 0;
  const int dyt[5] = {-1, -1, -1, 0, 0};
  const int dxt[5] = {-1,  0,  1, -1, 0};
  v8f a1 = {}, a2 = {}, a3 = {}, a4 = {};
  // per-lane weight fragment base; (tap,kc) step = 2048-half immediate offset
  size_t wb = (size_t)ntile * 512 + lane * 16;
  const bf16_t* pb1 = w1p + wb;
  const bf16_t* pb2 = w2p + wb;
  const bf16_t* pb3 = w3p + wb;
  const bf16_t* pb4 = w4p + wb;
  #pragma unroll
  for (int t = 0; t < NTAPS; ++t) {
    int slot = (1 + dyt[t]) * 34 + 1 + sub * 16 + m + dxt[t];
    const bf16_t* ap = lds + (size_t)slot * LDS_PITCH + kseg;
    #pragma unroll
    for (int kc = 0; kc < 2; ++kc) {
      const int fo = (t * 2 + kc) * 2048;   // compile-time immediate
      v16bf a   = load_a_frag(ap + kc * 32);
      v16bf bw1 = load_b_frag(pb1 + fo);
      v16bf bw2 = load_b_frag(pb2 + fo);
      v16bf bw3 = load_b_frag(pb3 + fo);
      v16bf bw4 = load_b_frag(pb4 + fo);
      a1 = wmma_bf16(a, bw1, a1);
      a2 = wmma_bf16(a, bw2, a2);
      a3 = wmma_bf16(a, bw3, a3);
      a4 = wmma_bf16(a, bw4, a4);
    }
  }

  int n = lane & 15, mb = (lane >> 4) * 8;
  int ch = ntile * 16 + n;
  float bb1 = b1[ch], bb2 = b2[ch], bb3 = b3[ch], bb4 = b4[ch];
  #pragma unroll
  for (int r = 0; r < 8; ++r) {
    int xloc = sub * 16 + mb + r;
    float hv = (float)lds[(size_t)(34 + 1 + xloc) * LDS_PITCH + ch];
    float g1 = sigmoidf_(a1[r] + bb1);
    float g2 = sigmoidf_(a2[r] + bb2);
    float g3 = sigmoidf_(a3[r] + bb3);
    float u  = tanhf(a4[r] + bb4);
    hout[(rowpix + x0 + xloc) * 64 + ch] = (bf16_t)(g1 * tanhf(g2 * hv + g3 * u));
  }
}

// ---------------------------------------------------------------------------
// d_in leaf order (JAX pytree, dict keys sorted):
//   0: inputs (2,4,32,128,128)   1: targets (2,4,1,128,128)
//   per RMB i (base = 2 + 20*i):
//     +0 b_in  +1 b_out
//     +2..+5  mu0 b1..b4   +6..+9  mu0 w1..w4
//     +10..13 mu1 b1..b4   +14..17 mu1 w1..w4
//     +18 w_in  +19 w_out
// ---------------------------------------------------------------------------
extern "C" void kernel_launch(void* const* d_in, const int* in_sizes, int n_in,
                              void* d_out, int out_size, void* d_ws, size_t ws_size,
                              hipStream_t stream) {
  (void)in_sizes; (void)n_in; (void)out_size; (void)ws_size;
  const float* inputs  = (const float*)d_in[0];
  const float* targets = (const float*)d_in[1];

  size_t off = 0;
  auto walloc = [&](size_t bytes) -> void* {
    void* p = (void*)((char*)d_ws + off);
    off += (bytes + 255) & ~(size_t)255;
    return p;
  };
  bf16_t* xin64 = (bf16_t*)walloc((size_t)NPIX * 64 * 2);
  bf16_t* x32a  = (bf16_t*)walloc((size_t)NPIX * 32 * 2);
  bf16_t* x32b  = (bf16_t*)walloc((size_t)NPIX * 32 * 2);
  bf16_t* hA    = (bf16_t*)walloc((size_t)NPIX * 64 * 2);
  bf16_t* hB    = (bf16_t*)walloc((size_t)NPIX * 64 * 2);

  bf16_t* wp_in[4];
  bf16_t* wp_mu[4][2][4];
  bf16_t* wp_out[4];
  for (int i = 0; i < 4; ++i) {
    int KpadIn = (i == 0) ? 64 : 32;
    wp_in[i] = (bf16_t*)walloc((size_t)(KpadIn / 32) * 4 * 512 * 2);
    for (int mm = 0; mm < 2; ++mm)
      for (int j = 0; j < 4; ++j)
        wp_mu[i][mm][j] = (bf16_t*)walloc((size_t)5 * 2 * 4 * 512 * 2);
    int O = (i == 3) ? 256 : 32;
    wp_out[i] = (bf16_t*)walloc((size_t)2 * (O / 16) * 512 * 2);
  }

  // ---- pack weights (every call; ws is not preserved) ----
  for (int i = 0; i < 4; ++i) {
    int base = 2 + i * 20;
    int Iin    = (i == 0) ? 33 : 32;
    int KpadIn = (i == 0) ? 64 : 32;
    {
      int total = (KpadIn / 32) * 4 * 512;
      pack_w_kernel<<<(total + 255) / 256, 256, 0, stream>>>(
          (const float*)d_in[base + 18], wp_in[i], 64, Iin, KpadIn, 0, 0, total);
    }
    for (int mm = 0; mm < 2; ++mm)
      for (int j = 0; j < 4; ++j) {
        const float* src = (const float*)d_in[base + 2 + mm * 8 + 4 + j];
        int total = 5 * 2 * 4 * 512;
        pack_w_kernel<<<(total + 255) / 256, 256, 0, stream>>>(
            src, wp_mu[i][mm][j], 64, 64, 64, 1, (i == 0) ? 1 : 0, total);
      }
    {
      int O = (i == 3) ? 256 : 32;
      int total = 2 * (O / 16) * 512;
      pack_w_kernel<<<(total + 255) / 256, 256, 0, stream>>>(
          (const float*)d_in[base + 19], wp_out[i], O, 64, 64, 0, 0, total);
    }
  }

  // ---- pack activations ----
  {
    int total = NPIX * 64;
    pack_input_kernel<<<(total + 255) / 256, 256, 0, stream>>>(inputs, targets,
                                                               xin64, x32a);
  }

  // ---- RMB stack ----
  bf16_t* xcur = x32a;
  bf16_t* xnext = x32b;
  const int TILES = NPIX / 16;
  const int MU_BLOCKS = NPIX / 32;   // 32 pixels per block
  for (int i = 0; i < 4; ++i) {
    int base = 2 + i * 20;
    const float* b_in  = (const float*)d_in[base + 0];
    const float* b_out = (const float*)d_in[base + 1];
    // 1x1 conv in -> hA (64ch): one wave covers all 4 ch-tiles
    {
      int ntasks = TILES;
      if (i == 0) {
        gemm1x1_kernel<4, 4, 2, 0><<<(ntasks + 7) / 8, 256, 0, stream>>>(
            xin64, 64, wp_in[i], b_in, nullptr, hA, nullptr, ntasks);
      } else {
        gemm1x1_kernel<4, 4, 1, 0><<<(ntasks + 7) / 8, 256, 0, stream>>>(
            xcur, 32, wp_in[i], b_in, nullptr, hA, nullptr, ntasks);
      }
    }
    // MUs (mask A: 4 taps, mask B: 5 taps)
    if (i == 0) {
      mu_kernel<4><<<MU_BLOCKS, 256, 0, stream>>>(
          hA, hB, wp_mu[i][0][0], wp_mu[i][0][1], wp_mu[i][0][2], wp_mu[i][0][3],
          (const float*)d_in[base + 2], (const float*)d_in[base + 3],
          (const float*)d_in[base + 4], (const float*)d_in[base + 5]);
      mu_kernel<4><<<MU_BLOCKS, 256, 0, stream>>>(
          hB, hA, wp_mu[i][1][0], wp_mu[i][1][1], wp_mu[i][1][2], wp_mu[i][1][3],
          (const float*)d_in[base + 10], (const float*)d_in[base + 11],
          (const float*)d_in[base + 12], (const float*)d_in[base + 13]);
    } else {
      mu_kernel<5><<<MU_BLOCKS, 256, 0, stream>>>(
          hA, hB, wp_mu[i][0][0], wp_mu[i][0][1], wp_mu[i][0][2], wp_mu[i][0][3],
          (const float*)d_in[base + 2], (const float*)d_in[base + 3],
          (const float*)d_in[base + 4], (const float*)d_in[base + 5]);
      mu_kernel<5><<<MU_BLOCKS, 256, 0, stream>>>(
          hB, hA, wp_mu[i][1][0], wp_mu[i][1][1], wp_mu[i][1][2], wp_mu[i][1][3],
          (const float*)d_in[base + 10], (const float*)d_in[base + 11],
          (const float*)d_in[base + 12], (const float*)d_in[base + 13]);
    }
    if (i < 3) {
      int ntasks = TILES;
      gemm1x1_kernel<2, 2, 2, 1><<<(ntasks + 7) / 8, 256, 0, stream>>>(
          hA, 64, wp_out[i], b_out, xcur, xnext, nullptr, ntasks);
      bf16_t* t = xcur; xcur = xnext; xnext = t;
    } else {
      int ntasks = TILES * 4;  // 4 waves/tile, 4 ch-tiles each
      gemm1x1_kernel<4, 16, 2, 2><<<(ntasks + 7) / 8, 256, 0, stream>>>(
          hA, 64, wp_out[i], b_out, nullptr, nullptr, (float*)d_out, ntasks);
    }
  }
}
